// DecisionTransformer_57045755625928
// MI455X (gfx1250) — matrix operand order repaired and checked by reading, not compiled, not run on hardware
//
#include <hip/hip_runtime.h>
#include <hip/hip_bf16.h>
#include <math.h>

// ---------------------------------------------------------------------------
// Decision Transformer forward on gfx1250 (MI455X), wave32 WMMA.
//   - weights pre-transposed/converted to f16 [N,K] (one-time pass, L2-resident)
//   - activations carried as f16 for matrix math, f32 for residual/LN stats
//   - all GEMMs + attention: v_wmma_f32_16x16x32_f16 through padded LDS tiles
// ---------------------------------------------------------------------------

typedef _Float16 v16h __attribute__((ext_vector_type(16)));
typedef _Float16 v8h  __attribute__((ext_vector_type(8)));
typedef float    v8f  __attribute__((ext_vector_type(8)));

#define BATCH 2
#define SEQ   512
#define T3    1536            // 3*SEQ tokens per batch element
#define NTOK  (BATCH * T3)    // 3072 rows total (batch concatenated)
#define CDIM  768
#define NHEAD 12
#define HDIM  64
#define FDIM  3072            // 4*C
#define NLAYER 6

// LDS strides (halves), padded so fragment chunks stay 16B/32B aligned.
#define AS_STRIDE 40          // 32+8  -> 80B rows  (v8h chunks 16B aligned)
#define BT_STRIDE 48          // 32+16 -> 96B rows  (v16h chunks 32B aligned)
#define KT_STRIDE 80          // 64+16 -> 160B rows (v16h chunks 32B aligned)

static __device__ __forceinline__ v16h make_v16(v8h lo, v8h hi) {
    return __builtin_shufflevector(lo, hi, 0, 1, 2, 3, 4, 5, 6, 7,
                                           8, 9, 10, 11, 12, 13, 14, 15);
}

static __device__ __forceinline__ float gelu_exact(float x) {
    return 0.5f * x * (1.0f + erff(x * 0.70710678118654752f));
}

// ---------------------------------------------------------------------------
// One-time weight transpose + f32->f16 convert:  W[K,N] -> Wt[N,K].
// grid (K/64, N/64, nmat); 256 threads; 64x64 tile via LDS.
// ---------------------------------------------------------------------------
__global__ __launch_bounds__(256) void wtrans_kernel(
    const float* __restrict__ W, _Float16* __restrict__ Wt, int K, int N)
{
    __shared__ _Float16 T[64][68];            // pad 4 halves against conflicts
    const size_t base = (size_t)blockIdx.z * (size_t)K * N;
    const int k0 = blockIdx.x * 64, n0 = blockIdx.y * 64;
    const int tid = threadIdx.x;
    const int r  = tid >> 2;                  // 0..63 (k within tile)
    const int cb = (tid & 3) * 16;            // 16 floats per thread

    #pragma unroll
    for (int u = 0; u < 4; ++u) {
        const float4 f = *(const float4*)&W[base + (size_t)(k0 + r) * N + n0 + cb + u * 4];
        T[r][cb + u * 4 + 0] = (_Float16)f.x;
        T[r][cb + u * 4 + 1] = (_Float16)f.y;
        T[r][cb + u * 4 + 2] = (_Float16)f.z;
        T[r][cb + u * 4 + 3] = (_Float16)f.w;
    }
    __syncthreads();

    const int n = tid >> 2;                   // 0..63 (n within tile)
    #pragma unroll
    for (int u = 0; u < 2; ++u) {
        const int ck = ((tid & 3) + u * 4) * 8;   // 0..56 (k chunk)
        v8h o;
        #pragma unroll
        for (int j = 0; j < 8; ++j) o[j] = T[ck + j][n];
        *(v8h*)&Wt[base + (size_t)(n0 + n) * K + k0 + ck] = o;
    }
}

// ---------------------------------------------------------------------------
// Embedding + interleave (z,s,a) + time embedding + LN0 -> X (f32) and Xh (f16).
// ---------------------------------------------------------------------------
__global__ __launch_bounds__(256) void embed_ln0_kernel(
    const float* __restrict__ states, const float* __restrict__ actions,
    const float* __restrict__ goals,  const int* __restrict__ timesteps,
    const float* __restrict__ W_s, const float* __restrict__ b_s,
    const float* __restrict__ W_a, const float* __restrict__ b_a,
    const float* __restrict__ W_g, const float* __restrict__ b_g,
    const float* __restrict__ E_t,
    const float* __restrict__ g0,  const float* __restrict__ bt0,
    float* __restrict__ X, _Float16* __restrict__ Xh)
{
    const int t    = blockIdx.x;              // 0..NTOK-1
    const int b    = t / T3;
    const int r    = t - b * T3;
    const int pos  = r / 3;
    const int kind = r - pos * 3;             // 0=z, 1=state, 2=action
    const int tid  = threadIdx.x;

    __shared__ float inv[64];
    __shared__ float yrow[CDIM];
    __shared__ float rs[256], rq[256];

    const float *src, *W, *bias;
    if (kind == 0)      { src = goals   + (size_t)b * 64;               W = W_g; bias = b_g; }
    else if (kind == 1) { src = states  + ((size_t)b * SEQ + pos) * 64; W = W_s; bias = b_s; }
    else                { src = actions + ((size_t)b * SEQ + pos) * 64; W = W_a; bias = b_a; }

    if (tid < 64) inv[tid] = src[tid];
    __syncthreads();

    const int ts = timesteps[b * SEQ + pos];
    const float* te = E_t + (size_t)ts * CDIM;

    float lsum = 0.f, lsq = 0.f;
    #pragma unroll
    for (int u = 0; u < 3; ++u) {
        const int c = tid + u * 256;
        float acc = bias[c] + te[c];
        #pragma unroll 8
        for (int k = 0; k < 64; ++k) acc += inv[k] * W[(size_t)k * CDIM + c];
        yrow[c] = acc;
        lsum += acc;
        lsq  += acc * acc;
    }
    rs[tid] = lsum; rq[tid] = lsq;
    __syncthreads();
    for (int s = 128; s > 0; s >>= 1) {
        if (tid < s) { rs[tid] += rs[tid + s]; rq[tid] += rq[tid + s]; }
        __syncthreads();
    }
    const float mean = rs[0] * (1.0f / CDIM);
    const float var  = rq[0] * (1.0f / CDIM) - mean * mean;
    const float rstd = rsqrtf(var + 1e-5f);
    float*     of = X  + (size_t)t * CDIM;
    _Float16*  oh = Xh + (size_t)t * CDIM;
    #pragma unroll
    for (int u = 0; u < 3; ++u) {
        const int c = tid + u * 256;
        const float v = (yrow[c] - mean) * rstd * g0[c] + bt0[c];
        of[c] = v;
        oh[c] = (_Float16)v;
    }
}

// ---------------------------------------------------------------------------
// Row LayerNorm -> f32 + f16 outputs. One 256-thread block per row.
// ---------------------------------------------------------------------------
__global__ __launch_bounds__(256) void ln_kernel(
    const float* __restrict__ In, const float* __restrict__ g,
    const float* __restrict__ b,
    float* __restrict__ OutF, _Float16* __restrict__ OutH)
{
    const int row = blockIdx.x;
    const int tid = threadIdx.x;
    const float* x = In + (size_t)row * CDIM;
    const float v0 = x[tid], v1 = x[tid + 256], v2 = x[tid + 512];

    __shared__ float rs[256], rq[256];
    rs[tid] = v0 + v1 + v2;
    rq[tid] = v0 * v0 + v1 * v1 + v2 * v2;
    __syncthreads();
    for (int s = 128; s > 0; s >>= 1) {
        if (tid < s) { rs[tid] += rs[tid + s]; rq[tid] += rq[tid + s]; }
        __syncthreads();
    }
    const float mean = rs[0] * (1.0f / CDIM);
    const float var  = rq[0] * (1.0f / CDIM) - mean * mean;
    const float rstd = rsqrtf(var + 1e-5f);
    float*    of = OutF + (size_t)row * CDIM;
    _Float16* oh = OutH + (size_t)row * CDIM;
    const float w0 = (v0 - mean) * rstd * g[tid]       + b[tid];
    const float w1 = (v1 - mean) * rstd * g[tid + 256] + b[tid + 256];
    const float w2 = (v2 - mean) * rstd * g[tid + 512] + b[tid + 512];
    of[tid] = w0;        of[tid + 256] = w1;        of[tid + 512] = w2;
    oh[tid] = (_Float16)w0; oh[tid + 256] = (_Float16)w1; oh[tid + 512] = (_Float16)w2;
}

// ---------------------------------------------------------------------------
// Tiled WMMA GEMM: Out = act( Ah[M,K] * Wt[N,K]^T + bias (+ Res) )
// Ah f16 row-major, Wt f16 N-major (pre-transposed). 64x64 block tile,
// K-step 32, 256 threads = 8 waves, 2 accumulators per wave.
// Writes f32 (OutF) and/or f16 (OutH). act: 0 none, 1 exact GELU.
// ---------------------------------------------------------------------------
__global__ __launch_bounds__(256) void gemm_wmma_kernel(
    const _Float16* __restrict__ Ah, const _Float16* __restrict__ Wt,
    const float* __restrict__ bias, const float* __restrict__ Res,
    float* __restrict__ OutF, _Float16* __restrict__ OutH,
    int M, int N, int K, int act)
{
    __shared__ __align__(32) _Float16 As[64 * AS_STRIDE]; // 64x32 A tile
    __shared__ __align__(32) _Float16 Bt[64 * BT_STRIDE]; // 64(n)x32(k) W tile

    const int bm   = blockIdx.x * 64;
    const int bn   = blockIdx.y * 64;
    const int tid  = threadIdx.x;
    const int lane = tid & 31;
    const int wave = tid >> 5;                 // 0..7
    const int wr   = (wave >> 1) * 16;         // wave row tile
    const int wc   = (wave & 1) * 32;          // wave col offset
    const int l16  = lane & 15;
    const bool hi  = lane >= 16;

    // Staging coordinates: each thread copies one 8-half (16B) chunk per tile.
    const int sr = tid >> 2;                   // 0..63 row
    const int sc = (tid & 3) * 8;              // 0/8/16/24 halves

    const _Float16* aG = Ah + (size_t)(bm + sr) * K + sc;
    const _Float16* bG = Wt + (size_t)(bn + sr) * K + sc;
    _Float16* aL = &As[sr * AS_STRIDE + sc];
    _Float16* bL = &Bt[sr * BT_STRIDE + sc];

    v8f acc0 = {}; v8f acc1 = {};

    for (int k0 = 0; k0 < K; k0 += 32) {
        __syncthreads();
        *(v8h*)aL = *(const v8h*)(aG + k0);    // global b128 -> ds b128
        *(v8h*)bL = *(const v8h*)(bG + k0);
        if (k0 + 32 < K) {
            __builtin_prefetch(aG + k0 + 32, 0, 1);
            __builtin_prefetch(bG + k0 + 32, 0, 1);
        }
        __syncthreads();

        // A fragment: row wr+l16; K halves off..off+15, off = hi?16:0.
        const _Float16* ar = &As[(wr + l16) * AS_STRIDE + (hi ? 16 : 0)];
        const v16h afrag = make_v16(*(const v8h*)ar, *(const v8h*)(ar + 8));

        // B fragments: lane column wc(+16)+l16; halves hi ? K16..31 : K0..15.
        const int boff = hi ? 16 : 0;
        const v16h bf0 = *(const v16h*)&Bt[(wc      + l16) * BT_STRIDE + boff];
        const v16h bf1 = *(const v16h*)&Bt[(wc + 16 + l16) * BT_STRIDE + boff];

        acc0 = __builtin_amdgcn_wmma_f32_16x16x32_f16(false, afrag, false, bf0,
                                                      (short)0, acc0, false, false);
        acc1 = __builtin_amdgcn_wmma_f32_16x16x32_f16(false, afrag, false, bf1,
                                                      (short)0, acc1, false, false);
    }

    // Epilogue: C/D layout -> row = j + (hi?8:0), col = l16.
    const int rof = hi ? 8 : 0;
    #pragma unroll
    for (int j = 0; j < 8; ++j) {
        const int row = bm + wr + rof + j;
        const int c0  = bn + wc + l16;
        const int c1  = c0 + 16;
        float v0 = acc0[j] + bias[c0];
        float v1 = acc1[j] + bias[c1];
        if (Res) {
            v0 += Res[(size_t)row * N + c0];
            v1 += Res[(size_t)row * N + c1];
        }
        if (act == 1) { v0 = gelu_exact(v0); v1 = gelu_exact(v1); }
        if (OutF) {
            OutF[(size_t)row * N + c0] = v0;
            OutF[(size_t)row * N + c1] = v1;
        }
        if (OutH) {
            OutH[(size_t)row * N + c0] = (_Float16)v0;
            OutH[(size_t)row * N + c1] = (_Float16)v1;
        }
    }
}

// ---------------------------------------------------------------------------
// Causal flash attention over f16 Q/K/V ([NTOK, CDIM], head h at cols h*64).
// One (batch*head, 64 q-rows) per 128-thread block (4 waves, 16 q-rows each).
// QK^T and P*V via WMMA; online softmax with 16-lane shfl_xor reductions.
// Scale 1/sqrt(D) applied in the softmax VALU path. Output f16.
// ---------------------------------------------------------------------------
__global__ __launch_bounds__(128) void attn_kernel(
    const _Float16* __restrict__ Qh, const _Float16* __restrict__ Kh,
    const _Float16* __restrict__ Vh, _Float16* __restrict__ Oh)
{
    __shared__ __align__(32) _Float16 Qs[64 * KT_STRIDE];   // Q tile
    __shared__ __align__(32) _Float16 Kt[64 * KT_STRIDE];   // Kt[key][d]
    __shared__ __align__(32) _Float16 Vt[64 * KT_STRIDE];   // Vt[d][key]
    __shared__ __align__(32) _Float16 Ps[64 * KT_STRIDE];   // P strips

    const int qb   = blockIdx.x;              // 0..23
    const int bh   = blockIdx.y;              // 0..23
    const int bb   = bh / NHEAD;
    const int h    = bh - bb * NHEAD;
    const int tid  = threadIdx.x;
    const int lane = tid & 31;
    const int wave = tid >> 5;                // 0..3
    const int l16  = lane & 15;
    const bool hi  = lane >= 16;
    const int wq   = wave * 16;
    const int rof  = hi ? 8 : 0;

    const size_t rowbase = (size_t)(bb * T3 + qb * 64);
    const int    hcol    = h * HDIM;

    // Stage Q tile: 512 x 16B chunks, 4 per thread.
    #pragma unroll
    for (int u = 0; u < 4; ++u) {
        const int idx = tid + u * 128;        // 0..511
        const int r = idx >> 3, c = (idx & 7) * 8;
        *(v8h*)&Qs[r * KT_STRIDE + c] =
            *(const v8h*)&Qh[(rowbase + r) * CDIM + hcol + c];
    }
    __syncthreads();

    // Preload the wave's two Q A-fragments (d = 0..31, 32..63).
    v16h aq[2];
    {
        const _Float16* ar = &Qs[(wq + l16) * KT_STRIDE];
        const int off = hi ? 16 : 0;
        #pragma unroll
        for (int s = 0; s < 2; ++s)
            aq[s] = make_v16(*(const v8h*)(ar + s * 32 + off),
                             *(const v8h*)(ar + s * 32 + off + 8));
    }

    v8f o0 = {}, o1 = {}, o2 = {}, o3 = {};
    float mrow[8], lrow[8];
    #pragma unroll
    for (int j = 0; j < 8; ++j) { mrow[j] = -3.0e38f; lrow[j] = 0.f; }

    for (int kb = 0; kb <= qb; ++kb) {
        __syncthreads();                       // prior iteration done with Kt/Vt
        const size_t kbase = (size_t)(bb * T3 + kb * 64);
        #pragma unroll
        for (int u = 0; u < 4; ++u) {
            const int idx = tid + u * 128;
            const int r = idx >> 3, c = (idx & 7) * 8;
            // K: straight b128 copy (B layout for QK^T wants Kt[key][d]).
            *(v8h*)&Kt[r * KT_STRIDE + c] =
                *(const v8h*)&Kh[(kbase + r) * CDIM + hcol + c];
            // V: b128 load + 8-way transpose scatter (Vt[d][key]).
            const v8h vv = *(const v8h*)&Vh[(kbase + r) * CDIM + hcol + c];
            #pragma unroll
            for (int j = 0; j < 8; ++j) Vt[(c + j) * KT_STRIDE + r] = vv[j];
        }
        __syncthreads();

        // S strip (16x64) = Q * K^T: 4 column tiles x 2 K-steps.
        v8f s0 = {}, s1 = {}, s2 = {}, s3 = {};
        #pragma unroll
        for (int s = 0; s < 2; ++s) {
            const int boff = s * 32 + (hi ? 16 : 0);
            const v16h b0 = *(const v16h*)&Kt[(     l16) * KT_STRIDE + boff];
            const v16h b1 = *(const v16h*)&Kt[(16 + l16) * KT_STRIDE + boff];
            const v16h b2 = *(const v16h*)&Kt[(32 + l16) * KT_STRIDE + boff];
            const v16h b3 = *(const v16h*)&Kt[(48 + l16) * KT_STRIDE + boff];
            s0 = __builtin_amdgcn_wmma_f32_16x16x32_f16(false, aq[s], false, b0, (short)0, s0, false, false);
            s1 = __builtin_amdgcn_wmma_f32_16x16x32_f16(false, aq[s], false, b1, (short)0, s1, false, false);
            s2 = __builtin_amdgcn_wmma_f32_16x16x32_f16(false, aq[s], false, b2, (short)0, s2, false, false);
            s3 = __builtin_amdgcn_wmma_f32_16x16x32_f16(false, aq[s], false, b3, (short)0, s3, false, false);
        }

        // Online softmax (scale folded in here).
        const bool diag = (kb == qb);
        #pragma unroll
        for (int j = 0; j < 8; ++j) {
            float e0 = s0[j] * 0.125f, e1 = s1[j] * 0.125f;
            float e2 = s2[j] * 0.125f, e3 = s3[j] * 0.125f;
            if (diag) {
                const int qloc = wq + rof + j;
                if (     l16 > qloc) e0 = -3.0e38f;
                if (16 + l16 > qloc) e1 = -3.0e38f;
                if (32 + l16 > qloc) e2 = -3.0e38f;
                if (48 + l16 > qloc) e3 = -3.0e38f;
            }
            float mj = fmaxf(fmaxf(e0, e1), fmaxf(e2, e3));
            #pragma unroll
            for (int d = 1; d < 16; d <<= 1) mj = fmaxf(mj, __shfl_xor(mj, d, 32));
            const float mnew = fmaxf(mrow[j], mj);
            const float corr = __expf(mrow[j] - mnew);
            const float p0 = __expf(e0 - mnew);
            const float p1 = __expf(e1 - mnew);
            const float p2 = __expf(e2 - mnew);
            const float p3 = __expf(e3 - mnew);
            float sum = p0 + p1 + p2 + p3;
            #pragma unroll
            for (int d = 1; d < 16; d <<= 1) sum += __shfl_xor(sum, d, 32);
            lrow[j] = lrow[j] * corr + sum;
            mrow[j] = mnew;
            o0[j] *= corr; o1[j] *= corr; o2[j] *= corr; o3[j] *= corr;
            _Float16* pr = &Ps[(wq + rof + j) * KT_STRIDE];
            pr[     l16] = (_Float16)p0;
            pr[16 + l16] = (_Float16)p1;
            pr[32 + l16] = (_Float16)p2;
            pr[48 + l16] = (_Float16)p3;
        }
        __syncthreads();                       // publish P, order vs Vt reads

        // O strip += P (16x64) * V: A-fragments from Ps, B from Vt.
        const _Float16* prow = &Ps[(wq + l16) * KT_STRIDE];
        const int aoff = hi ? 16 : 0;
        #pragma unroll
        for (int s = 0; s < 2; ++s) {
            const v16h ap = make_v16(*(const v8h*)(prow + s * 32 + aoff),
                                     *(const v8h*)(prow + s * 32 + aoff + 8));
            const int boff = s * 32 + (hi ? 16 : 0);
            const v16h b0 = *(const v16h*)&Vt[(     l16) * KT_STRIDE + boff];
            const v16h b1 = *(const v16h*)&Vt[(16 + l16) * KT_STRIDE + boff];
            const v16h b2 = *(const v16h*)&Vt[(32 + l16) * KT_STRIDE + boff];
            const v16h b3 = *(const v16h*)&Vt[(48 + l16) * KT_STRIDE + boff];
            o0 = __builtin_amdgcn_wmma_f32_16x16x32_f16(false, ap, false, b0, (short)0, o0, false, false);
            o1 = __builtin_amdgcn_wmma_f32_16x16x32_f16(false, ap, false, b1, (short)0, o1, false, false);
            o2 = __builtin_amdgcn_wmma_f32_16x16x32_f16(false, ap, false, b2, (short)0, o2, false, false);
            o3 = __builtin_amdgcn_wmma_f32_16x16x32_f16(false, ap, false, b3, (short)0, o3, false, false);
        }
    }

    // Normalize and write f16 attention output.
    #pragma unroll
    for (int j = 0; j < 8; ++j) {
        const float inv = 1.0f / lrow[j];
        _Float16* op = Oh + (rowbase + wq + rof + j) * CDIM + hcol;
        op[     l16] = (_Float16)(o0[j] * inv);
        op[16 + l16] = (_Float16)(o1[j] * inv);
        op[32 + l16] = (_Float16)(o2[j] * inv);
        op[48 + l16] = (_Float16)(o3[j] * inv);
    }
}

// ---------------------------------------------------------------------------
// Output heads. One block per (b, pos). Output layout [2, B, SEQ, 64].
// ---------------------------------------------------------------------------
__global__ __launch_bounds__(128) void heads_kernel(
    const float* __restrict__ X,
    const float* __restrict__ W_ps, const float* __restrict__ b_ps,
    const float* __restrict__ W_pa, const float* __restrict__ b_pa,
    float* __restrict__ out)
{
    const int bp  = blockIdx.x;               // 0..B*SEQ-1
    const int b   = bp >> 9;
    const int pos = bp & 511;
    const int tid = threadIdx.x;

    __shared__ float r1[CDIM];                // state-token row  (kind 1)
    __shared__ float r2[CDIM];                // action-token row (kind 2)
    const size_t t1 = (size_t)b * T3 + (size_t)pos * 3 + 1;
    #pragma unroll
    for (int u = 0; u < 6; ++u) {
        const int c = tid + u * 128;
        r1[c] = X[t1 * CDIM + c];
        r2[c] = X[(t1 + 1) * CDIM + c];
    }
    __syncthreads();

    if (tid < 64) {
        float acc = b_ps[tid];
        for (int k = 0; k < CDIM; ++k) acc += r2[k] * W_ps[(size_t)k * 64 + tid];
        out[((size_t)(0 * BATCH + b) * SEQ + pos) * 64 + tid] = acc;
    } else {
        const int c = tid - 64;
        float acc = b_pa[c];
        for (int k = 0; k < CDIM; ++k) acc += r1[k] * W_pa[(size_t)k * 64 + c];
        out[((size_t)(1 * BATCH + b) * SEQ + pos) * 64 + c] = tanhf(acc);
    }
}

// ---------------------------------------------------------------------------
// Host launcher.
// ---------------------------------------------------------------------------
extern "C" void kernel_launch(void* const* d_in, const int* in_sizes, int n_in,
                              void* d_out, int out_size, void* d_ws, size_t ws_size,
                              hipStream_t stream) {
    (void)in_sizes; (void)n_in; (void)out_size; (void)ws_size;

    const float* states    = (const float*)d_in[0];
    const float* actions   = (const float*)d_in[1];
    const float* goals     = (const float*)d_in[2];
    /* d_in[3] = traj_mask (unused) */
    const int*   timesteps = (const int*)  d_in[4];
    const float* W_s  = (const float*)d_in[5];
    const float* b_s  = (const float*)d_in[6];
    const float* W_a  = (const float*)d_in[7];
    const float* b_a  = (const float*)d_in[8];
    const float* W_g  = (const float*)d_in[9];
    const float* b_g  = (const float*)d_in[10];
    const float* E_t  = (const float*)d_in[11];
    const float* ln0g = (const float*)d_in[12];
    const float* ln0b = (const float*)d_in[13];
    const float* Wq   = (const float*)d_in[14];
    const float* bq   = (const float*)d_in[15];
    const float* Wk   = (const float*)d_in[16];
    const float* bk   = (const float*)d_in[17];
    const float* Wv   = (const float*)d_in[18];
    const float* bv   = (const float*)d_in[19];
    const float* Wp   = (const float*)d_in[20];
    const float* bp   = (const float*)d_in[21];
    const float* ln1g = (const float*)d_in[22];
    const float* ln1b = (const float*)d_in[23];
    const float* W1   = (const float*)d_in[24];
    const float* b1   = (const float*)d_in[25];
    const float* W2   = (const float*)d_in[26];
    const float* b2   = (const float*)d_in[27];
    const float* ln2g = (const float*)d_in[28];
    const float* ln2b = (const float*)d_in[29];
    const float* W_ps = (const float*)d_in[30];
    const float* b_ps = (const float*)d_in[31];
    const float* W_pa = (const float*)d_in[32];
    const float* b_pa = (const float*)d_in[33];

    // ---------------- workspace carve-up (byte offsets) ----------------
    char* ws8 = (char*)d_ws;
    const size_t SZ_F32 = (size_t)NTOK * CDIM * sizeof(float);      // 9.44 MB
    const size_t SZ_H   = (size_t)NTOK * CDIM * sizeof(_Float16);   // 4.72 MB
    const size_t SZ_HH  = (size_t)NTOK * FDIM * sizeof(_Float16);   // 18.9 MB
    const size_t SZ_WSQ = (size_t)NLAYER * CDIM * CDIM * sizeof(_Float16);
    const size_t SZ_WF  = (size_t)NLAYER * CDIM * FDIM * sizeof(_Float16);

    size_t off = 0;
    float*    X   = (float*)   (ws8 + off); off += SZ_F32;
    float*    XB  = (float*)   (ws8 + off); off += SZ_F32;
    _Float16* Xh  = (_Float16*)(ws8 + off); off += SZ_H;
    _Float16* Qh  = (_Float16*)(ws8 + off); off += SZ_H;
    _Float16* Kh  = (_Float16*)(ws8 + off); off += SZ_H;
    _Float16* Vh  = (_Float16*)(ws8 + off); off += SZ_H;
    _Float16* Ath = (_Float16*)(ws8 + off); off += SZ_H;
    _Float16* Hh  = (_Float16*)(ws8 + off); off += SZ_HH;
    _Float16* Wqt = (_Float16*)(ws8 + off); off += SZ_WSQ;
    _Float16* Wkt = (_Float16*)(ws8 + off); off += SZ_WSQ;
    _Float16* Wvt = (_Float16*)(ws8 + off); off += SZ_WSQ;
    _Float16* Wpt = (_Float16*)(ws8 + off); off += SZ_WSQ;
    _Float16* W1t = (_Float16*)(ws8 + off); off += SZ_WF;
    _Float16* W2t = (_Float16*)(ws8 + off); off += SZ_WF;

    // ---------------- one-time weight transpose/convert ----------------
    const dim3 gSQ(CDIM / 64, CDIM / 64, NLAYER);   // 12 x 12 x 6
    const dim3 gW1(CDIM / 64, FDIM / 64, NLAYER);   // 12 x 48 x 6
    const dim3 gW2(FDIM / 64, CDIM / 64, NLAYER);   // 48 x 12 x 6
    wtrans_kernel<<<gSQ, 256, 0, stream>>>(Wq, Wqt, CDIM, CDIM);
    wtrans_kernel<<<gSQ, 256, 0, stream>>>(Wk, Wkt, CDIM, CDIM);
    wtrans_kernel<<<gSQ, 256, 0, stream>>>(Wv, Wvt, CDIM, CDIM);
    wtrans_kernel<<<gSQ, 256, 0, stream>>>(Wp, Wpt, CDIM, CDIM);
    wtrans_kernel<<<gW1, 256, 0, stream>>>(W1, W1t, CDIM, FDIM);
    wtrans_kernel<<<gW2, 256, 0, stream>>>(W2, W2t, FDIM, CDIM);

    // ---------------- forward pass ----------------
    embed_ln0_kernel<<<NTOK, 256, 0, stream>>>(
        states, actions, goals, timesteps,
        W_s, b_s, W_a, b_a, W_g, b_g, E_t, ln0g, ln0b, X, Xh);

    const dim3 gCC(NTOK / 64, CDIM / 64);     // 48 x 12
    const dim3 gCF(NTOK / 64, FDIM / 64);     // 48 x 48
    const dim3 gAtt(T3 / 64, BATCH * NHEAD);  // 24 x 24

    for (int l = 0; l < NLAYER; ++l) {
        const size_t wo = (size_t)l * CDIM * CDIM;
        const size_t fo = (size_t)l * CDIM * FDIM;
        const float* bq_l = bq + (size_t)l * CDIM;
        const float* bk_l = bk + (size_t)l * CDIM;
        const float* bv_l = bv + (size_t)l * CDIM;
        const float* bp_l = bp + (size_t)l * CDIM;
        const float* b1_l = b1 + (size_t)l * FDIM;
        const float* b2_l = b2 + (size_t)l * CDIM;

        gemm_wmma_kernel<<<gCC, 256, 0, stream>>>(Xh, Wqt + wo, bq_l, nullptr,
                                                  nullptr, Qh, NTOK, CDIM, CDIM, 0);
        gemm_wmma_kernel<<<gCC, 256, 0, stream>>>(Xh, Wkt + wo, bk_l, nullptr,
                                                  nullptr, Kh, NTOK, CDIM, CDIM, 0);
        gemm_wmma_kernel<<<gCC, 256, 0, stream>>>(Xh, Wvt + wo, bv_l, nullptr,
                                                  nullptr, Vh, NTOK, CDIM, CDIM, 0);
        attn_kernel<<<gAtt, 128, 0, stream>>>(Qh, Kh, Vh, Ath);
        gemm_wmma_kernel<<<gCC, 256, 0, stream>>>(Ath, Wpt + wo, bp_l, X,
                                                  XB, nullptr, NTOK, CDIM, CDIM, 0);
        ln_kernel<<<NTOK, 256, 0, stream>>>(XB, ln1g + (size_t)l * CDIM,
                                            ln1b + (size_t)l * CDIM, X, Xh);
        gemm_wmma_kernel<<<gCF, 256, 0, stream>>>(Xh, W1t + fo, b1_l, nullptr,
                                                  nullptr, Hh, NTOK, FDIM, CDIM, 1);
        gemm_wmma_kernel<<<gCC, 256, 0, stream>>>(Hh, W2t + fo, b2_l, X,
                                                  XB, nullptr, NTOK, CDIM, FDIM, 0);
        ln_kernel<<<NTOK, 256, 0, stream>>>(XB, ln2g + (size_t)l * CDIM,
                                            ln2b + (size_t)l * CDIM, X, Xh);
    }

    heads_kernel<<<BATCH * SEQ, 128, 0, stream>>>(X, W_ps, b_ps, W_pa, b_pa,
                                                  (float*)d_out);
}
